// SJN_Meta_2_19705309954371
// MI455X (gfx1250) — compile-verified, matrix operand order checked by reading, not run on hardware
//
#include <hip/hip_runtime.h>

// ---------------------------------------------------------------------------
// GNN MetaLayer x3 forward for MI455X (gfx1250, wave32, WMMA + TDM).
// - BatchNorm folded into per-stage weights (prep kernel): each stage is one
//   streaming f16-WMMA GEMM pass with f32 accumulate.
// - Inter-stage activations stored in f16 (halves dominant HBM traffic).
// - Next-stage batch statistics fused into GEMM epilogue (shfl+LDS+atomics).
// - Tile staging for pass-through f16 stages uses the Tensor Data Mover
//   (tensor_load_to_lds, per-wave 16-row descriptor, TENSORcnt wait),
//   guarded by __has_builtin with a manual-staging fallback.
//   This toolchain (clang-23 / amdgpu-toolchain) uses the 6-arg builtin:
//   (uint32x4 g0, int32x8 g1, int32x4 g2, int32x4 g3, int32x8 g4, i32 cpol).
// ---------------------------------------------------------------------------

typedef __attribute__((ext_vector_type(16))) _Float16 v16h;
typedef __attribute__((ext_vector_type(8)))  float    v8f;
typedef __attribute__((ext_vector_type(4)))  unsigned v4u;
typedef __attribute__((ext_vector_type(8)))  int      v8i;
typedef __attribute__((ext_vector_type(4)))  int      v4i;

#define N_NODES 50000
#define N_EDGES 800000
#define STATS_RPB 1024

#if defined(__HIP_DEVICE_COMPILE__) && __has_builtin(__builtin_amdgcn_tensor_load_to_lds) && __has_builtin(__builtin_amdgcn_s_wait_tensorcnt)
#define HAVE_TDM 1
#else
#define HAVE_TDM 0
#endif

// -------------------------- device helpers --------------------------------

__device__ __forceinline__ float seg_one(const void* p, int w, const int* idx,
                                         int f16, int R, int c)
{
    int r = idx ? idx[R] : R;
    long long o = (long long)r * w + c;
    if (f16) return (float)((const _Float16*)p)[o];
    return ((const float*)p)[o];
}

__device__ __forceinline__ float seg_load(int R, int c,
    const void* p0, int w0, const int* i0, int f0,
    const void* p1, int w1, const int* i1, int f1,
    const void* p2, int w2, const int* i2, int f2)
{
    if (c < w0) return seg_one(p0, w0, i0, f0, R, c);
    c -= w0;
    if (c < w1) return seg_one(p1, w1, i1, f1, R, c);
    c -= w1;
    if (c < w2) return seg_one(p2, w2, i2, f2, R, c);
    return 0.0f;
}

// ---------------- standalone column stats (gather-concat inputs) -----------

__global__ __launch_bounds__(256)
void stats_kernel(const void* p0, int w0, const int* i0, int f0,
                  const void* p1, int w1, const int* i1, int f1,
                  const void* p2, int w2, const int* i2, int f2,
                  int Ktot, int M, float* __restrict__ stats)
{
    __shared__ float s0[96];
    __shared__ float s1[96];
    int t = threadIdx.x;
    for (int i = t; i < 192; i += 256) { if (i < 96) s0[i] = 0.f; else s1[i-96] = 0.f; }
    __syncthreads();

    int base = blockIdx.x * STATS_RPB;
    int rows = M - base; if (rows > STATS_RPB) rows = STATS_RPB;
    int total = rows * Ktot;
    for (int idx = t; idx < total; idx += 256) {
        int r = idx / Ktot;
        int c = idx - r * Ktot;
        float v = seg_load(base + r, c, p0,w0,i0,f0, p1,w1,i1,f1, p2,w2,i2,f2);
        atomicAdd(&s0[c], v);
        atomicAdd(&s1[c], v * v);
    }
    __syncthreads();
    for (int c = t; c < Ktot; c += 256) {
        atomicAdd(&stats[c],      s0[c]);
        atomicAdd(&stats[96 + c], s1[c]);
    }
}

// ------------- fold BN into weights: W' = diag(a)W (f16), bias' ------------

__global__ __launch_bounds__(256)
void prep_kernel(const float* __restrict__ stats,
                 const float* __restrict__ g, const float* __restrict__ beta,
                 const float* __restrict__ W, const float* __restrict__ bias,
                 int Ktot, int KPAD, int dout, int NTpad, float invM,
                 _Float16* __restrict__ Wp, float* __restrict__ biasP)
{
    __shared__ float sa[96];
    __shared__ float sb[96];
    int t = threadIdx.x;
    if (t < Ktot) {
        float mean = stats[t] * invM;
        float var  = stats[96 + t] * invM - mean * mean;
        float rs   = rsqrtf(var + 1e-5f);
        float a    = g[t] * rs;
        sa[t] = a;
        sb[t] = beta[t] - mean * a;
    }
    __syncthreads();
    for (int j = t; j < dout; j += 256) {
        float s = bias[j];
        for (int k = 0; k < Ktot; ++k) s += sb[k] * W[k * dout + j];
        biasP[j] = s;
    }
    int tot = NTpad * KPAD;
    for (int idx = t; idx < tot; idx += 256) {
        int n = idx / KPAD;
        int k = idx - n * KPAD;
        float v = (n < dout && k < Ktot) ? sa[k] * W[k * dout + n] : 0.0f;
        Wp[idx] = (_Float16)v;
    }
}

// ----------------------------- WMMA GEMM -----------------------------------
// out[M x dout] = act[M x Ktot] @ Wp^T + biasP, optional LeakyReLU, optional
// fused column sum/sumsq of the post-activation output.
// Block: 256 thr = 8 waves; each wave owns 16 rows, all NCT 16-col tiles.
// TDMIN: input is a single dense f16 matrix of width 64 == KPAD; stage the
// 128x64 tile with tensor_load_to_lds (one 16-row descriptor per wave).

template<int KPAD, int NCT, bool LEAKY, bool OUTF16, bool TDMIN>
__global__ __launch_bounds__(256)
void gemm_kernel(const void* p0, int w0, const int* i0, int f0,
                 const void* p1, int w1, const int* i1, int f1,
                 const void* p2, int w2, const int* i2, int f2,
                 int Ktot, int M, int dout,
                 const _Float16* __restrict__ Wp,
                 const float* __restrict__ biasP,
                 void* __restrict__ outv,
                 float* statsOut)
{
    __shared__ _Float16 tile[128 * KPAD];
    __shared__ float ls[96];
    __shared__ float lss[96];

    int t = threadIdx.x;
    int tileBase = blockIdx.x * 128;
    int lane = t & 31;
    int wave = t >> 5;

    if (t < 96) { ls[t] = 0.f; lss[t] = 0.f; }

#if HAVE_TDM
    constexpr bool USE_TDM = TDMIN;
#else
    constexpr bool USE_TDM = false;
#endif

    if constexpr (USE_TDM) {
#if HAVE_TDM
        // Per-wave TDM: 16 rows x 64 f16 (2KB) slice of the 128x64 tile.
        int start = tileBase + wave * 16;
        unsigned long long ga = (unsigned long long)p0 + (unsigned long long)start * 128ull;
        unsigned lds = (unsigned)(size_t)(void*)tile + (unsigned)wave * 2048u;
        int dim1 = M - start; if (dim1 < 0) dim1 = 0;

        v4u g0;
        g0.x = 1u;                                            // count=1, user desc
        g0.y = lds;                                           // lds_addr
        g0.z = (unsigned)ga;                                  // global_addr[31:0]
        g0.w = (unsigned)((ga >> 32) & 0x01FFFFFFu) | 0x80000000u; // [56:32] | type=2
        v8i g1;
        g1[0] = 0x00010000;                                   // data_size=2B
        g1[1] = (int)(64u << 16);                             // tensor_dim0=64
        g1[2] = (int)(((unsigned)dim1 & 0xFFFFu) << 16);      // tensor_dim1 lo
        g1[3] = (int)((((unsigned)dim1 >> 16) & 0xFFFFu) | (64u << 16)); // dim1 hi | tile_dim0=64
        g1[4] = 16;                                           // tile_dim1=16 rows
        g1[5] = 64;                                           // tensor_dim0_stride=64
        g1[6] = 0;
        g1[7] = 0;
        v4i g2 = (v4i){0, 0, 0, 0};
        v4i g3 = (v4i){0, 0, 0, 0};
        v8i g4 = (v8i){0, 0, 0, 0, 0, 0, 0, 0};
        __builtin_amdgcn_tensor_load_to_lds(g0, g1, g2, g3, g4, 0);
        __builtin_amdgcn_s_wait_tensorcnt((short)0);
#endif
    } else {
        // Manual staging: f32/f16 gather-concat -> f16 tile, zero padded.
        for (int idx = t; idx < 128 * KPAD; idx += 256) {
            int r = idx / KPAD;
            int c = idx - r * KPAD;
            int R = tileBase + r;
            float v = 0.0f;
            if (R < M && c < Ktot)
                v = seg_load(R, c, p0,w0,i0,f0, p1,w1,i1,f1, p2,w2,i2,f2);
            tile[idx] = (_Float16)v;
        }
    }
    __syncthreads();

    int mrow = lane & 15;
    int kh   = lane >> 4;
    int ncol = lane & 15;

    v8f acc[NCT];
#pragma unroll
    for (int ct = 0; ct < NCT; ++ct)
        acc[ct] = (v8f){0.f,0.f,0.f,0.f,0.f,0.f,0.f,0.f};

    const _Float16* rowA = &tile[(wave * 16 + mrow) * KPAD];

#pragma unroll
    for (int kt = 0; kt < KPAD / 32; ++kt) {
        union { v16h v; unsigned u[8]; } A;
#pragma unroll
        for (int j = 0; j < 8; ++j) {
            int kk = (j < 4) ? (2*j + 8*kh) : (16 + 2*(j-4) + 8*kh);
            A.u[j] = *(const unsigned*)(rowA + kt*32 + kk);
        }
#pragma unroll
        for (int ct = 0; ct < NCT; ++ct) {
            union { v16h v; unsigned u[8]; } B;
            const _Float16* rowB = Wp + (ct*16 + ncol) * KPAD + kt*32;
#pragma unroll
            for (int j = 0; j < 8; ++j) {
                int kk = (j < 4) ? (2*j + 8*kh) : (16 + 2*(j-4) + 8*kh);
                B.u[j] = *(const unsigned*)(rowB + kk);
            }
            acc[ct] = __builtin_amdgcn_wmma_f32_16x16x32_f16(
                false, A.v, false, B.v, (short)0, acc[ct], false, false);
        }
    }

    // Epilogue: bias + LeakyReLU + store + fused next-stage column stats
#pragma unroll
    for (int ct = 0; ct < NCT; ++ct) {
        int n = ct * 16 + ncol;
        float psum = 0.f, psq = 0.f;
        if (n < dout) {
            float bv = biasP[n];
#pragma unroll
            for (int v = 0; v < 8; ++v) {
                int R = tileBase + wave * 16 + v + 8 * kh;
                if (R < M) {
                    float val = acc[ct][v] + bv;
                    if (LEAKY) val = (val > 0.f) ? val : 0.1f * val;
                    long long o = (long long)R * dout + n;
                    if (OUTF16) ((_Float16*)outv)[o] = (_Float16)val;
                    else        ((float*)outv)[o]    = val;
                    psum += val;
                    psq  += val * val;
                }
            }
        }
        if (statsOut) {
            psum += __shfl_xor(psum, 16, 32);
            psq  += __shfl_xor(psq,  16, 32);
            if (lane < 16 && n < dout) {
                atomicAdd(&ls[n],  psum);
                atomicAdd(&lss[n], psq);
            }
        }
    }
    __syncthreads();
    if (statsOut) {
        for (int i = t; i < 192; i += 256)
            atomicAdd(&statsOut[i], (i < 96) ? ls[i] : lss[i - 96]);
    }
}

// --------------------------- scatter-sum -----------------------------------

__global__ __launch_bounds__(256)
void scatter_kernel(const _Float16* __restrict__ m, const int* __restrict__ colIdx,
                    float* __restrict__ agg, int total)
{
    int idx = blockIdx.x * 256 + threadIdx.x;
    if (idx < total) {
        int e = idx >> 6;
        int c = idx & 63;
        atomicAdd(&agg[(long long)colIdx[e] * 64 + c], (float)m[idx]);
    }
}

// ------------------------- final linear + sigmoid ---------------------------

__global__ __launch_bounds__(256)
void pred_kernel(const void* __restrict__ feat, int isF16, int M, int F,
                 const float* __restrict__ W, const float* __restrict__ b,
                 float* __restrict__ out)
{
    int i = blockIdx.x * 256 + threadIdx.x;
    if (i < M) {
        float s = b[0];
        for (int k = 0; k < F; ++k) {
            float v = isF16 ? (float)((const _Float16*)feat)[(long long)i * F + k]
                            : ((const float*)feat)[(long long)i * F + k];
            s += v * W[k];
        }
        out[i] = 1.0f / (1.0f + expf(-s));
    }
}

// ------------------------------ host side ----------------------------------

struct Seg { const void* p; int w; const int* idx; int f16; };

static void run_stage(hipStream_t st, Seg a, Seg b, Seg c,
                      int Ktot, int dout, int M, int variant,
                      const float* W, const float* beta, const float* bias, const float* g,
                      float* statsIn, bool standaloneStats, float* statsOut,
                      _Float16* Wp, float* biasP, void* out)
{
    static const int KPADv[6] = {64,64,64,64,96,96};
    static const int NCTv[6]  = {4,4,2,4,5,1};
    int KPAD = KPADv[variant], NCT = NCTv[variant];

    if (standaloneStats) {
        (void)hipMemsetAsync(statsIn, 0, 192 * sizeof(float), st);
        int grid = (M + STATS_RPB - 1) / STATS_RPB;
        stats_kernel<<<grid, 256, 0, st>>>(a.p,a.w,a.idx,a.f16, b.p,b.w,b.idx,b.f16,
                                           c.p,c.w,c.idx,c.f16, Ktot, M, statsIn);
    }
    prep_kernel<<<1, 256, 0, st>>>(statsIn, g, beta, W, bias,
                                   Ktot, KPAD, dout, NCT * 16, 1.0f / (float)M,
                                   Wp, biasP);
    if (statsOut) (void)hipMemsetAsync(statsOut, 0, 192 * sizeof(float), st);

    dim3 grid((M + 127) / 128);
#define GEMM_GO(KP,NC,LK,OF,TD) gemm_kernel<KP,NC,LK,OF,TD><<<grid,256,0,st>>>( \
        a.p,a.w,a.idx,a.f16, b.p,b.w,b.idx,b.f16, c.p,c.w,c.idx,c.f16, \
        Ktot, M, dout, Wp, biasP, out, statsOut)
    switch (variant) {
        case 0: GEMM_GO(64,4,true ,true ,false); break;
        case 1: GEMM_GO(64,4,true ,true ,true ); break;
        case 2: GEMM_GO(64,2,false,true ,true ); break;
        case 3: GEMM_GO(64,4,false,true ,true ); break;
        case 4: GEMM_GO(96,5,true ,false,false); break;
        default:GEMM_GO(96,1,false,false,false); break;
    }
#undef GEMM_GO
}

extern "C" void kernel_launch(void* const* d_in, const int* in_sizes, int n_in,
                              void* d_out, int out_size, void* d_ws, size_t ws_size,
                              hipStream_t stream)
{
    (void)n_in; (void)out_size; (void)ws_size;
    const int N = N_NODES, E = N_EDGES;

    // ---- input layout: detect flattening order via in_sizes[0] ----
    const float *x0; const void* ea0; const int *eidx; int base;
    if (in_sizes[0] == N) {
        // fully-sorted tree flatten: batch, edge_attr, edge_index, params..., x
        ea0  = d_in[1];
        eidx = (const int*)d_in[2];
        base = 3;
        x0   = (const float*)d_in[base + 2 + 108 + 2];
    } else {
        // top-level insertion order: x, edge_index, edge_attr, batch, params...
        x0   = (const float*)d_in[0];
        eidx = (const int*)  d_in[1];
        ea0  = d_in[2];
        base = 4;
    }
    const float* elinW = (const float*)d_in[base + 0];
    const float* elinB = (const float*)d_in[base + 1];
    const float* xlinW = (const float*)d_in[base + 2 + 108];
    const float* xlinB = (const float*)d_in[base + 2 + 108 + 1];
    // stage params (sorted keys inside each stage dict): W, beta, bias, g
    auto P = [&](int layer, int mlp, int s, int which) -> const float* {
        return (const float*)d_in[base + 2 + layer * 36 + mlp * 12 + s * 4 + which];
    };
    const int* row = eidx;
    const int* col = eidx + E;

    // ---- workspace carve ----
    char* w = (char*)d_ws;
    size_t off = 0;
    auto alloc = [&](size_t bytes) -> void* {
        void* p = w + off;
        off = (off + bytes + 255) & ~(size_t)255;
        return p;
    };
    _Float16* hE_a  = (_Float16*)alloc((size_t)E * 64 * 2);   // f16 intermediates
    _Float16* hE_b  = (_Float16*)alloc((size_t)E * 64 * 2);
    _Float16* eaA   = (_Float16*)alloc((size_t)E * 19 * 2);
    _Float16* eaB   = (_Float16*)alloc((size_t)E * 19 * 2);
    float*    hN_a  = (float*)   alloc((size_t)N * 80 * 4);
    float*    hN_b  = (float*)   alloc((size_t)N * 80 * 4);
    float*    xA    = (float*)   alloc((size_t)N * 16 * 4);
    float*    xB    = (float*)   alloc((size_t)N * 16 * 4);
    float*    agg   = (float*)   alloc((size_t)N * 64 * 4);
    float*    stA   = (float*)   alloc(192 * 4);
    float*    stB   = (float*)   alloc(192 * 4);
    _Float16* Wp    = (_Float16*)alloc((size_t)80 * 96 * 2);
    float*    biasP = (float*)   alloc(80 * 4);

    const void* xcur  = x0;     int xf16 = 0;   // layer-0 x is f32 input
    const void* eacur = ea0;    int eaf16 = 0;  // layer-0 ea is f32 input
    float*    xbuf[2]  = { xA, xB };
    _Float16* eabuf[2] = { eaA, eaB };

    for (int l = 0; l < 3; ++l) {
        _Float16* eanext = eabuf[l & 1];
        float*    xnext  = xbuf[l & 1];
        Seg nil = { nullptr, 0, nullptr, 0 };

        // ---- Edge MLP: [x[row]|x[col]|ea] (54) -> 64 -> 64 -> 19 ----
        run_stage(stream, {xcur,16,row,xf16}, {xcur,16,col,xf16}, {eacur,19,nullptr,eaf16},
                  54, 64, E, 0,
                  P(l,0,0,0), P(l,0,0,1), P(l,0,0,2), P(l,0,0,3),
                  stA, true, stB, Wp, biasP, hE_a);
        run_stage(stream, {hE_a,64,nullptr,1}, nil, nil,
                  64, 64, E, 1,
                  P(l,0,1,0), P(l,0,1,1), P(l,0,1,2), P(l,0,1,3),
                  stB, false, stA, Wp, biasP, hE_b);
        run_stage(stream, {hE_b,64,nullptr,1}, nil, nil,
                  64, 19, E, 2,
                  P(l,0,2,0), P(l,0,2,1), P(l,0,2,2), P(l,0,2,3),
                  stA, false, nullptr, Wp, biasP, eanext);

        // ---- Node MLP1: [x[row]|ea'] (35) -> 64 -> 64 -> 64 ----
        run_stage(stream, {xcur,16,row,xf16}, {eanext,19,nullptr,1}, nil,
                  35, 64, E, 0,
                  P(l,1,0,0), P(l,1,0,1), P(l,1,0,2), P(l,1,0,3),
                  stA, true, stB, Wp, biasP, hE_a);
        run_stage(stream, {hE_a,64,nullptr,1}, nil, nil,
                  64, 64, E, 1,
                  P(l,1,1,0), P(l,1,1,1), P(l,1,1,2), P(l,1,1,3),
                  stB, false, stA, Wp, biasP, hE_b);
        run_stage(stream, {hE_b,64,nullptr,1}, nil, nil,
                  64, 64, E, 3,
                  P(l,1,2,0), P(l,1,2,1), P(l,1,2,2), P(l,1,2,3),
                  stA, false, nullptr, Wp, biasP, hE_a);   // m -> hE_a (f16)

        // ---- scatter-sum: agg[col] += m ----
        (void)hipMemsetAsync(agg, 0, (size_t)N * 64 * 4, stream);
        {
            int total = E * 64;
            scatter_kernel<<<(total + 255) / 256, 256, 0, stream>>>(hE_a, col, agg, total);
        }

        // ---- Node MLP2: [x|agg] (80) -> 80 -> 80 -> 16 ----
        run_stage(stream, {xcur,16,nullptr,xf16}, {agg,64,nullptr,0}, nil,
                  80, 80, N, 4,
                  P(l,2,0,0), P(l,2,0,1), P(l,2,0,2), P(l,2,0,3),
                  stA, true, stB, Wp, biasP, hN_a);
        run_stage(stream, {hN_a,80,nullptr,0}, nil, nil,
                  80, 80, N, 4,
                  P(l,2,1,0), P(l,2,1,1), P(l,2,1,2), P(l,2,1,3),
                  stB, false, stA, Wp, biasP, hN_b);
        run_stage(stream, {hN_b,80,nullptr,0}, nil, nil,
                  80, 16, N, 5,
                  P(l,2,2,0), P(l,2,2,1), P(l,2,2,2), P(l,2,2,3),
                  stA, false, nullptr, Wp, biasP, xnext);

        xcur  = xnext;  xf16  = 0;
        eacur = eanext; eaf16 = 1;
    }

    // ---- heads: sigmoid(x@W+b), sigmoid(ea@W+b); d_out = [y(N) | e(E)] ----
    float* outp = (float*)d_out;
    pred_kernel<<<(N + 255) / 256, 256, 0, stream>>>(xcur,  0, N, 16, xlinW, xlinB, outp);
    pred_kernel<<<(E + 255) / 256, 256, 0, stream>>>(eacur, 1, E, 19, elinW, elinB, outp + N);
}
